// PatchRelevance_45981919871429
// MI455X (gfx1250) — matrix operand order repaired and checked by reading, not compiled, not run on hardware
//
#include <hip/hip_runtime.h>

typedef _Float16 f16;
typedef __attribute__((ext_vector_type(16))) _Float16 v16h;
typedef __attribute__((ext_vector_type(8)))  _Float16 v8h;
typedef __attribute__((ext_vector_type(8)))  float    v8f;
typedef __attribute__((ext_vector_type(4)))  float    v4f;

#define N_TOK   16384
#define IN_DIM  256
#define OUT_DIM 128
#define NTILE   (N_TOK / 16)     // 1024 q row-tiles
#define SPLIT   4                // key-range splits (flash split-K)

__device__ __forceinline__ v16h combine16(v8h lo, v8h hi) {
    v16h r;
#pragma unroll
    for (int i = 0; i < 8; ++i) { r[i] = lo[i]; r[i + 8] = hi[i]; }
    return r;
}

__device__ __forceinline__ v8f wmma_f16(v16h a, v16h b, v8f c) {
    // D = A(16x32 f16) * B(32x16 f16) + C(16x16 f32)
    return __builtin_amdgcn_wmma_f32_16x16x32_f16(false, a, false, b,
                                                  (short)0, c, false, false);
}

// ---------------------------------------------------------------------------
// Transpose+convert W[256][128] f32 -> Wt[128][256] f16 (one-shot, 32K elems)
// so projection B-fragments are pairs of contiguous 8-half runs.
// ---------------------------------------------------------------------------
__global__ __launch_bounds__(256)
void TransWKernel_45981919871429(const float* __restrict__ W,
                                 f16* __restrict__ Wt) {
    const int idx = blockIdx.x * 256 + threadIdx.x;   // idx = c*256 + k
    const int k = idx & (IN_DIM - 1);
    const int c = idx >> 8;
    Wt[idx] = (f16)W[(size_t)k * OUT_DIM + c];
}

// ---------------------------------------------------------------------------
// QKV projection: one wave32 per 16-row tile. D[16x128] = x[16x256] * W + b.
// Wt is the f16 transposed weight [128][256].
// transposed==0 -> dst row-major f16 [N, 128]   (Q, K)
// transposed==1 -> dst f16 [128, N] (V^T: makes P*V B-fragments contiguous)
// ---------------------------------------------------------------------------
__global__ __launch_bounds__(32)
void ProjKernel_45981919871429(const float* __restrict__ x,
                               const f16* __restrict__ Wt,
                               const float* __restrict__ bias,
                               f16* __restrict__ dst, int transposed) {
    const int lane = threadIdx.x;
    const int n = lane & 15;      // B-col / A-row within tile
    const int g = lane >> 4;      // half-wave group
    const int r0 = blockIdx.x * 16;

    // A fragments of x (f32 -> f16), 8 k-blocks of 32.
    v16h ax[8];
#pragma unroll
    for (int kb = 0; kb < 8; ++kb) {
        const float* p = x + (size_t)(r0 + n) * IN_DIM + kb * 32;
        v4f a0 = *(const v4f*)(p + 8 * g);
        v4f a1 = *(const v4f*)(p + 8 * g + 4);
        v4f a2 = *(const v4f*)(p + 16 + 8 * g);
        v4f a3 = *(const v4f*)(p + 16 + 8 * g + 4);
        v16h a;
#pragma unroll
        for (int i = 0; i < 4; ++i) {
            a[i]      = (f16)a0[i];
            a[4 + i]  = (f16)a1[i];
            a[8 + i]  = (f16)a2[i];
            a[12 + i] = (f16)a3[i];
        }
        ax[kb] = a;
    }

#pragma unroll
    for (int t = 0; t < 8; ++t) {               // 8 output column tiles of 16
        v8f acc = {};
#pragma unroll
        for (int kb = 0; kb < 8; ++kb) {
            // B element e -> K = kb*32 + 8g + e (e<8), kb*32 + 16 + 8g + (e-8) (e>=8), col = t*16 + n
            const f16* wp = Wt + (size_t)(t * 16 + n) * IN_DIM + kb * 32 + 8 * g;
            v16h b = combine16(*(const v8h*)wp, *(const v8h*)(wp + 16));
            acc = wmma_f16(ax[kb], b, acc);
        }
        const float bv = bias[t * 16 + n];
        if (!transposed) {
#pragma unroll
            for (int v = 0; v < 8; ++v)
                dst[(size_t)(r0 + v + 8 * g) * OUT_DIM + t * 16 + n] =
                    (f16)(acc[v] + bv);
        } else {
#pragma unroll
            for (int v = 0; v < 8; ++v)
                dst[(size_t)(t * 16 + n) * N_TOK + (r0 + v + 8 * g)] =
                    (f16)(acc[v] + bv);
        }
    }
}

// ---------------------------------------------------------------------------
// Flash attention partial pass: one wave32 per (16-row q tile, key split).
// Streams 32 key columns/step: 8 WMMAs for S = Q*K^T, online softmax,
// 8 WMMAs for O += P*V. Writes UNNORMALIZED partials (O, m, l).
// ---------------------------------------------------------------------------
__global__ __launch_bounds__(32)
void AttnKernel_45981919871429(const f16* __restrict__ Qh,
                               const f16* __restrict__ Kh,
                               const f16* __restrict__ Vt,
                               float* __restrict__ Opart,
                               float* __restrict__ mpart,
                               float* __restrict__ lpart) {
    __shared__ alignas(16) f16 Pl[16 * 32];     // P tile relayout staging
    const int lane = threadIdx.x;
    const int n = lane & 15;
    const int g = lane >> 4;
    const int tile = blockIdx.x;
    const int split = blockIdx.y;
    const int r0 = tile * 16;
    const int jbeg = split * (N_TOK / SPLIT);
    const int jend = jbeg + (N_TOK / SPLIT);

    // Loop-invariant Q A-fragments (D=128 -> 4 k-blocks of 32)
    v16h aq[4];
#pragma unroll
    for (int kb = 0; kb < 4; ++kb) {
        const f16* p = Qh + (size_t)(r0 + n) * OUT_DIM + kb * 32;
        aq[kb] = combine16(*(const v8h*)(p + 8 * g),
                           *(const v8h*)(p + 16 + 8 * g));
    }

    float m_s[8], l_s[8];
    v8f o[8];
#pragma unroll
    for (int v = 0; v < 8; ++v) { m_s[v] = -__builtin_inff(); l_s[v] = 0.f; }
#pragma unroll
    for (int t = 0; t < 8; ++t) { v8f z = {}; o[t] = z; }

    for (int j0 = jbeg; j0 < jend; j0 += 32) {
        // ---- S = Q * K^T, 2 column sub-tiles ----
        v8f S[2];
#pragma unroll
        for (int s = 0; s < 2; ++s) {
            v8f acc = {};
#pragma unroll
            for (int kb = 0; kb < 4; ++kb) {
                const f16* kp = Kh + (size_t)(j0 + 16 * s + n) * OUT_DIM
                                   + kb * 32 + 8 * g;
                v16h b = combine16(*(const v8h*)kp, *(const v8h*)(kp + 16));
                acc = wmma_f16(aq[kb], b, acc);
            }
            S[s] = acc;
        }

        // ---- online softmax over this 32-column chunk ----
        float mx[8];
#pragma unroll
        for (int v = 0; v < 8; ++v) mx[v] = fmaxf(S[0][v], S[1][v]);
#pragma unroll
        for (int off = 1; off < 16; off <<= 1)
#pragma unroll
            for (int v = 0; v < 8; ++v)
                mx[v] = fmaxf(mx[v], __shfl_xor(mx[v], off, 32));

        float sc[8];
#pragma unroll
        for (int v = 0; v < 8; ++v) {
            const float mn = fmaxf(m_s[v], mx[v]);
            sc[v] = __expf(m_s[v] - mn);
            m_s[v] = mn;
            l_s[v] *= sc[v];
        }
#pragma unroll
        for (int t = 0; t < 8; ++t)
#pragma unroll
            for (int v = 0; v < 8; ++v) o[t][v] *= sc[v];

        float rs[8];
#pragma unroll
        for (int v = 0; v < 8; ++v) {
            const float p0 = __expf(S[0][v] - m_s[v]);
            const float p1 = __expf(S[1][v] - m_s[v]);
            rs[v] = p0 + p1;
            Pl[(v + 8 * g) * 32 + n]      = (f16)p0;
            Pl[(v + 8 * g) * 32 + 16 + n] = (f16)p1;
        }
#pragma unroll
        for (int off = 1; off < 16; off <<= 1)
#pragma unroll
            for (int v = 0; v < 8; ++v) rs[v] += __shfl_xor(rs[v], off, 32);
#pragma unroll
        for (int v = 0; v < 8; ++v) l_s[v] += rs[v];

        __syncthreads();   // single-wave WG: HW NOP, forces LDS waits/ordering

        // ---- O += P * V ----
        v16h ap = combine16(*(const v8h*)(&Pl[n * 32 + 8 * g]),
                            *(const v8h*)(&Pl[n * 32 + 16 + 8 * g]));
#pragma unroll
        for (int t = 0; t < 8; ++t) {
            const f16* vp = Vt + (size_t)(t * 16 + n) * N_TOK + j0 + 8 * g;
            v16h b = combine16(*(const v8h*)vp, *(const v8h*)(vp + 16));
            o[t] = wmma_f16(ap, b, o[t]);
        }
        __syncthreads();   // WAR on Pl vs next iteration
    }

    // ---- write unnormalized partials ----
    float* Ob = Opart + ((size_t)(split * NTILE + tile) * 16) * OUT_DIM;
#pragma unroll
    for (int t = 0; t < 8; ++t)
#pragma unroll
        for (int v = 0; v < 8; ++v)
            Ob[(v + 8 * g) * OUT_DIM + t * 16 + n] = o[t][v];

    if (n == 0) {
        float* mb = mpart + (size_t)(split * NTILE + tile) * 16;
        float* lb = lpart + (size_t)(split * NTILE + tile) * 16;
#pragma unroll
        for (int v = 0; v < 8; ++v) {
            mb[v + 8 * g] = m_s[v];
            lb[v + 8 * g] = l_s[v];
        }
    }
}

// ---------------------------------------------------------------------------
// Combine the SPLIT partials: O = (sum_s exp(m_s - M) O_s) / (sum_s exp(m_s-M) l_s)
// ---------------------------------------------------------------------------
__global__ __launch_bounds__(128)
void CombineKernel_45981919871429(const float* __restrict__ Opart,
                                  const float* __restrict__ mpart,
                                  const float* __restrict__ lpart,
                                  float* __restrict__ out) {
    const int tile = blockIdx.x;
    const int c = threadIdx.x;          // 0..127 output column
    for (int r = 0; r < 16; ++r) {
        float mv[SPLIT];
        float M = -__builtin_inff();
#pragma unroll
        for (int s = 0; s < SPLIT; ++s) {
            mv[s] = mpart[(size_t)(s * NTILE + tile) * 16 + r];
            M = fmaxf(M, mv[s]);
        }
        float L = 0.f, acc = 0.f;
#pragma unroll
        for (int s = 0; s < SPLIT; ++s) {
            const float f = __expf(mv[s] - M);
            L += f * lpart[(size_t)(s * NTILE + tile) * 16 + r];
            acc += f * Opart[((size_t)(s * NTILE + tile) * 16 + r) * OUT_DIM + c];
        }
        out[(size_t)(tile * 16 + r) * OUT_DIM + c] = acc / L;
    }
}

extern "C" void kernel_launch(void* const* d_in, const int* in_sizes, int n_in,
                              void* d_out, int out_size, void* d_ws, size_t ws_size,
                              hipStream_t stream) {
    const float* x  = (const float*)d_in[0];
    const float* Wq = (const float*)d_in[1];
    const float* bq = (const float*)d_in[2];
    const float* Wk = (const float*)d_in[3];
    const float* bk = (const float*)d_in[4];
    const float* Wv = (const float*)d_in[5];
    const float* bv = (const float*)d_in[6];
    float* out = (float*)d_out;

    // ---- workspace carve-up (all buffers fully rewritten every launch) ----
    char* ws = (char*)d_ws;
    f16* Qh  = (f16*)ws;                    ws += (size_t)N_TOK * OUT_DIM * sizeof(f16); // 4 MB
    f16* Kh  = (f16*)ws;                    ws += (size_t)N_TOK * OUT_DIM * sizeof(f16); // 4 MB
    f16* Vt  = (f16*)ws;                    ws += (size_t)N_TOK * OUT_DIM * sizeof(f16); // 4 MB
    f16* Wtq = (f16*)ws;                    ws += (size_t)IN_DIM * OUT_DIM * sizeof(f16);
    f16* Wtk = (f16*)ws;                    ws += (size_t)IN_DIM * OUT_DIM * sizeof(f16);
    f16* Wtv = (f16*)ws;                    ws += (size_t)IN_DIM * OUT_DIM * sizeof(f16);
    float* mpart = (float*)ws;              ws += (size_t)SPLIT * N_TOK * sizeof(float);
    float* lpart = (float*)ws;              ws += (size_t)SPLIT * N_TOK * sizeof(float);
    float* Opart = (float*)ws;              // SPLIT * N * 128 f32 = 32 MB

    const dim3 blk32(32);
    const dim3 grdTile(NTILE);

    TransWKernel_45981919871429<<<dim3((IN_DIM * OUT_DIM) / 256), 256, 0, stream>>>(Wq, Wtq);
    TransWKernel_45981919871429<<<dim3((IN_DIM * OUT_DIM) / 256), 256, 0, stream>>>(Wk, Wtk);
    TransWKernel_45981919871429<<<dim3((IN_DIM * OUT_DIM) / 256), 256, 0, stream>>>(Wv, Wtv);

    ProjKernel_45981919871429<<<grdTile, blk32, 0, stream>>>(x, Wtq, bq, Qh, 0);
    ProjKernel_45981919871429<<<grdTile, blk32, 0, stream>>>(x, Wtk, bk, Kh, 0);
    ProjKernel_45981919871429<<<grdTile, blk32, 0, stream>>>(x, Wtv, bv, Vt, 1);

    AttnKernel_45981919871429<<<dim3(NTILE, SPLIT), blk32, 0, stream>>>(
        Qh, Kh, Vt, Opart, mpart, lpart);

    CombineKernel_45981919871429<<<grdTile, 128, 0, stream>>>(
        Opart, mpart, lpart, out);
}
